// GCN_18193481466251
// MI455X (gfx1250) — matrix operand order repaired and checked by reading, not compiled, not run on hardware
//
#include <hip/hip_runtime.h>
#include <hip/hip_bf16.h>

// ---------------------------------------------------------------------------
// CDNA5 (gfx1250) implementation of the text-GCN reference.
// All GEMM-like work goes through v_wmma_f32_16x16x32_f16 (wave32 WMMA).
// B matrices are pre-swizzled into WMMA fragment order so the hot loop does
// pure b128 loads; GEMM waves use 2x2 register blocking (4 WMMAs per
// 2 A-fragment + 2 B-fragment loads).
// ---------------------------------------------------------------------------

typedef __attribute__((ext_vector_type(16))) _Float16 v16h;
typedef __attribute__((ext_vector_type(8)))  float    v8f;

#define BB     32
#define NW     500
#define NSENT  50
#define SL     40
#define EMB    300
#define EMBP   320       // EMB padded to multiple of 32 (WMMA K)
#define HID    150
#define HIDP   160       // HID padded to multiple of 16/32
#define G4     600       // 4*HID
#define G4P    608       // padded to multiple of 16
#define NN     1600      // BB*NSENT  (sentence nodes)
#define NNODE  17600     // BB*(NW+NSENT)
#define EDGES  563200

#define CDIV(a, b) (((a) + (b) - 1) / (b))

// ------------------------------- helpers -----------------------------------

__device__ __forceinline__ float pos_enc(int p, int c) {
  // sinusoid table row p, col c (d=300). Row 0 is the zeroed padding row.
  if (p == 0) return 0.f;
  float ang = (float)p * __expf(-(float)(2 * (c >> 1)) * (9.210340371976184f / 300.f));
  return (c & 1) ? __cosf(ang) : __sinf(ang);
}

// A fragment: 16x32 f16, per ISA 7.12.2. pa is this lane's base pointer
// (row (lane&15), col offset (lane>=16 ? 8 : 0)); reads pa[k..k+7], pa[k+16..k+23].
__device__ __forceinline__ v16h load_a_frag(const _Float16* __restrict__ pa, int k) {
  v16h a;
#pragma unroll
  for (int e = 0; e < 8; ++e) a[e] = pa[k + e];
#pragma unroll
  for (int e = 0; e < 8; ++e) a[8 + e] = pa[k + 16 + e];
  return a;
}

// B fragment from fragment-major packed buffer: 16 contiguous halves per lane.
__device__ __forceinline__ v16h load_b_frag(const _Float16* __restrict__ pb) {
  v16h b;
#pragma unroll
  for (int e = 0; e < 16; ++e) b[e] = pb[e];
  return b;
}

__device__ __forceinline__ v8f wmma_f16(v16h a, v16h b, v8f c) {
  return __builtin_amdgcn_wmma_f32_16x16x32_f16(false, a, false, b, (short)0, c, false, false);
}

// --------------------------- WMMA GEMM kernels ------------------------------

// C[M,Np] = A[M,K](f16) x Bf(fragment-packed f16 [K/32][Np/16][32][16]) + bias.
// M % 32 == 0, K % 32 == 0, Np % 32 == 0.  One wave per 32x32 C block
// (2x2 tiles: 4 WMMAs reuse 2 A-fragments and 2 B-fragments per k-step).
__global__ void k_wmma_gemm(const _Float16* __restrict__ A, const _Float16* __restrict__ Bf,
                            float* __restrict__ C, const float* __restrict__ bias,
                            int M, int K, int Np, int N0) {
  int lane   = threadIdx.x & 31;
  int wave   = threadIdx.x >> 5;
  int ntiles = Np >> 4;
  int nb2    = ntiles >> 1;
  int mb2    = M >> 5;
  int blk    = blockIdx.x * (blockDim.x >> 5) + wave;
  if (blk >= mb2 * nb2) return;                      // wave-uniform: EXEC stays all-1
  int mt2 = blk / nb2, nt2 = blk - mt2 * nb2;
  const _Float16* pa0 = A + (mt2 * 32 + (lane & 15)) * K + ((lane >> 4) << 3);
  const _Float16* pa1 = pa0 + 16 * K;
  const _Float16* pb0 = Bf + (size_t)(nt2 * 64 + lane) * 16;
  const _Float16* pb1 = pb0 + 512;                   // next N tile (32 lanes * 16)
  const int bstep = ntiles << 9;                     // ntiles*32*16 halves per ktile
  v8f c00 = {}, c01 = {}, c10 = {}, c11 = {};
#pragma unroll 2
  for (int k = 0; k < K; k += 32) {
    v16h a0 = load_a_frag(pa0, k);
    v16h a1 = load_a_frag(pa1, k);
    v16h b0 = load_b_frag(pb0);
    v16h b1 = load_b_frag(pb1);
    pb0 += bstep; pb1 += bstep;
    c00 = wmma_f16(a0, b0, c00);
    c01 = wmma_f16(a0, b1, c01);
    c10 = wmma_f16(a1, b0, c10);
    c11 = wmma_f16(a1, b1, c11);
  }
  int n = lane & 15, mb = (lane >> 4) << 3;
  int col0 = nt2 * 32 + n;
  int col1 = col0 + 16;
  float bb0 = (bias != nullptr && col0 < N0) ? bias[col0] : 0.f;
  float bb1 = (bias != nullptr && col1 < N0) ? bias[col1] : 0.f;
  float* Crow0 = C + (mt2 * 32 + mb) * Np;
  float* Crow1 = Crow0 + 16 * Np;
#pragma unroll
  for (int r = 0; r < 8; ++r) {
    Crow0[r * Np + col0] = c00[r] + bb0;
    Crow0[r * Np + col1] = c01[r] + bb1;
    Crow1[r * Np + col0] = c10[r] + bb0;
    Crow1[r * Np + col1] = c11[r] + bb1;
  }
}

// CNN n-gram conv for window height h: per sentence an im2col GEMM
// [positions x h*EMBP] x fragment-packed [h*EMBP x 64(pad of 50)], then
// relu+max over positions via integer atomicMax on non-negative bit patterns.
// 1x2 N-blocking: the A fragment is reused across both output-channel tiles.
__global__ void k_conv(const _Float16* __restrict__ enc, const _Float16* __restrict__ Wf,
                       const float* __restrict__ bias, unsigned int* __restrict__ feats, int h) {
  int lane = threadIdx.x & 31;
  int wave = threadIdx.x >> 5;
  int tile = blockIdx.x * (blockDim.x >> 5) + wave;
  if (tile >= NN * 6) return;                        // 3 mtiles x 2 nblocks per sentence
  int n    = tile / 6;
  int rem  = tile - n * 6;
  int mt   = rem >> 1;
  int ntl2 = rem & 1;
  int Mpos = SL + 1 - h;                             // valid output positions (41-h)
  if (mt * 16 >= Mpos) return;                       // wave-uniform
  int K = h * EMBP;
  // overlapping im2col rows: lda = EMBP (consecutive positions), contiguous tokens
  const _Float16* pa  = enc + (n * SL + mt * 16 + (lane & 15)) * EMBP + ((lane >> 4) << 3);
  const _Float16* pb0 = Wf + (size_t)(ntl2 * 64 + lane) * 16;
  const _Float16* pb1 = pb0 + 512;
  const int bstep = 4 << 9;                          // 4 ntiles * 512
  v8f c0 = {}, c1 = {};
#pragma unroll 2
  for (int k = 0; k < K; k += 32) {
    v16h a  = load_a_frag(pa, k);
    v16h b0 = load_b_frag(pb0);
    v16h b1 = load_b_frag(pb1);
    pb0 += bstep; pb1 += bstep;
    c0 = wmma_f16(a, b0, c0);
    c1 = wmma_f16(a, b1, c1);
  }
  int nn = lane & 15, mb = (lane >> 4) << 3;
#pragma unroll
  for (int half = 0; half < 2; ++half) {
    int o = ntl2 * 32 + half * 16 + nn;
    if (o >= 50) continue;
    float bo = bias[o];
    float vmax = 0.f;
#pragma unroll
    for (int r = 0; r < 8; ++r) {
      int row = mt * 16 + mb + r;
      float v = (half ? c1[r] : c0[r]) + bo;
      if (row < Mpos) vmax = fmaxf(vmax, v);
    }
    vmax = fmaxf(vmax, 0.f);                         // relu; max(relu)=relu(max)>=0
    atomicMax(&feats[n * EMB + (h - 2) * 50 + o], __float_as_uint(vmax));
  }
}

// ---------------------------- element kernels -------------------------------

__global__ void k_embed_words(const int* __restrict__ x, const float* __restrict__ tab,
                              float* __restrict__ out) {
  int gid = blockIdx.x * blockDim.x + threadIdx.x;
  if (gid >= BB * NW * EMB) return;
  int w = gid / EMB, c = gid - w * EMB;
  out[gid] = tab[x[w] * EMB + c];
}

__global__ void k_lens(const int* __restrict__ sent_x, int* __restrict__ sentlen,
                       int* __restrict__ glen) {
  int gid = blockIdx.x * blockDim.x + threadIdx.x;
  if (gid < NN) {
    int cnt = 0;
    for (int l = 0; l < SL; ++l) cnt += (sent_x[gid * SL + l] != 0) ? 1 : 0;
    sentlen[gid] = cnt;
  }
  if (gid < BB) {
    int cnt = 0;
    for (int s = 0; s < NSENT; ++s) cnt += (sent_x[(gid * NSENT + s) * SL] != 0) ? 1 : 0;
    glen[gid] = cnt;
  }
}

// enc[n,l,c] = embed[tok] + pos_tab[input_pos]  (f16, padded to EMBP)
__global__ void k_enc(const int* __restrict__ sent_x, const float* __restrict__ tab,
                      const int* __restrict__ sentlen, _Float16* __restrict__ enc) {
  int gid = blockIdx.x * blockDim.x + threadIdx.x;
  if (gid >= NN * SL * EMBP) return;
  int c  = gid % EMBP;
  int nl = gid / EMBP;
  int l  = nl % SL;
  int n  = nl / SL;
  float v = 0.f;
  if (c < EMB) {
    int tok = sent_x[nl];
    int p   = (l < sentlen[n]) ? (l + 1) : 0;
    v = tab[tok * EMB + c] + pos_enc(p, c);
  }
  enc[gid] = (_Float16)v;
}

// conv weight [50,1,h,EMB] -> f16 fragment-packed B [h*EMBP/32][4][32][16]
__global__ void k_pack_convw(const float* __restrict__ w, _Float16* __restrict__ dst, int h) {
  int gid = blockIdx.x * blockDim.x + threadIdx.x;
  if (gid >= h * EMBP * 64) return;
  int e    = gid & 15;
  int lane = (gid >> 4) & 31;
  int frag = gid >> 9;                 // kt*4 + nt
  int kt = frag >> 2, nt = frag & 3;
  int o  = nt * 16 + (lane & 15);
  int kk = kt * 32 + ((lane >> 4) << 3) + (e & 7) + ((e >> 3) << 4);       // im2col K index
  int k = kk / EMBP, c = kk - k * EMBP;
  float v = (c < EMB && o < 50) ? w[(o * h + k) * EMB + c] : 0.f;
  dst[gid] = (_Float16)v;
}

// ngram (+sentence-node pos enc) -> f16 [NN, EMBP]
__global__ void k_ngram(const unsigned int* __restrict__ feats, _Float16* __restrict__ dst) {
  int gid = blockIdx.x * blockDim.x + threadIdx.x;
  if (gid >= NN * EMBP) return;
  int c = gid % EMBP, n = gid / EMBP;
  float v = 0.f;
  if (c < EMB) v = __uint_as_float(feats[n * EMB + c]) + pos_enc(n % NSENT, c);
  dst[gid] = (_Float16)v;
}

// W[N0,K0] (row-major) -> B = W^T fragment-packed f16 [KP/32][NP/16][32][16]
__global__ void k_pack_wT(const float* __restrict__ W, _Float16* __restrict__ dst,
                          int N0, int K0, int KP, int NP) {
  int gid = blockIdx.x * blockDim.x + threadIdx.x;
  if (gid >= KP * NP) return;
  int e      = gid & 15;
  int lane   = (gid >> 4) & 31;
  int frag   = gid >> 9;               // kt*ntiles + nt
  int ntiles = NP >> 4;
  int kt = frag / ntiles, nt = frag - kt * ntiles;
  int n = nt * 16 + (lane & 15);
  int k = kt * 32 + ((lane >> 4) << 3) + (e & 7) + ((e >> 3) << 4);
  float v = (k < K0 && n < N0) ? W[n * K0 + k] : 0.f;
  dst[gid] = (_Float16)v;
}

// f32 [M, srcld] -> f16 [M, KP], cols >= K0 zeroed
__global__ void k_pack_a(const float* __restrict__ src, _Float16* __restrict__ dst,
                         int M, int K0, int srcld, int KP) {
  int gid = blockIdx.x * blockDim.x + threadIdx.x;
  if (gid >= M * KP) return;
  int m = gid / KP, k = gid - m * KP;
  dst[gid] = (_Float16)((k < K0) ? src[m * srcld + k] : 0.f);
}

__global__ void k_addbias(const float* __restrict__ a, const float* __restrict__ b,
                          float* __restrict__ out, int n) {
  int gid = blockIdx.x * blockDim.x + threadIdx.x;
  if (gid < n) out[gid] = a[gid] + b[gid];
}

// One block per (batch b, direction d): full 50-step LSTM recurrence.
// xpart holds x@W_ih^T + (b_ih+b_hh) for all (b,step) rows, [2][NN][G4P].
// Reverse dir implements packed-sequence rev_within via bijective index map.
__global__ void k_lstm(const float* __restrict__ xpart, const float* __restrict__ Whh,
                       const int* __restrict__ glen, float* __restrict__ out) {
  int b = blockIdx.x;
  int d = blockIdx.y;
  int tid = threadIdx.x;
  __shared__ float hsh[HID];
  __shared__ float gsh[G4];
  if (tid < HID) hsh[tid] = 0.f;
  float creg = 0.f;
  __syncthreads();
  int gl = glen[b];
  const float* W  = Whh + d * G4 * HID;
  const float* xp = xpart + d * NN * G4P;
  for (int t = 0; t < NSENT; ++t) {
    int idx = (d == 1 && t < gl) ? (gl - 1 - t) : t;
    int row = b * NSENT + idx;
    if (tid < G4) {
      const float* wr = W + tid * HID;
      float s = xp[row * G4P + tid];
#pragma unroll 6
      for (int j = 0; j < HID; ++j) s = fmaf(hsh[j], wr[j], s);
      gsh[tid] = s;
    }
    __syncthreads();
    if (tid < HID) {
      float ig = 1.f / (1.f + __expf(-gsh[tid]));
      float fg = 1.f / (1.f + __expf(-gsh[HID + tid]));
      float gg = tanhf(gsh[2 * HID + tid]);
      float og = 1.f / (1.f + __expf(-gsh[3 * HID + tid]));
      creg = fg * creg + ig * gg;
      float hv = og * tanhf(creg);
      hsh[tid] = hv;
      out[row * EMBP + d * HID + tid] = hv;
    }
    __syncthreads();
  }
}

// masked h2 (zero rows with s >= glen[b]) -> f16 [NN, EMBP]
__global__ void k_pack_mask(const float* __restrict__ h2, const int* __restrict__ glen,
                            _Float16* __restrict__ dst) {
  int gid = blockIdx.x * blockDim.x + threadIdx.x;
  if (gid >= NN * EMBP) return;
  int c = gid % EMBP, n = gid / EMBP;
  int b = n / NSENT, s = n - b * NSENT;
  float v = 0.f;
  if (c < 2 * HID && s < glen[b]) v = h2[n * EMBP + c];
  dst[gid] = (_Float16)v;
}

// nodes = concat([emb_words, concat(cnn_feat, lstm_feat)], axis=1) -> f16 [NNODE, EMBP]
__global__ void k_nodes(const float* __restrict__ embw, const float* __restrict__ cp,
                        const float* __restrict__ lf, _Float16* __restrict__ dst) {
  int gid = blockIdx.x * blockDim.x + threadIdx.x;
  if (gid >= NNODE * EMBP) return;
  int c  = gid % EMBP;
  int nn = gid / EMBP;
  int b = nn / (NW + NSENT), r = nn - b * (NW + NSENT);
  float v = 0.f;
  if (c < EMB) {
    if (r < NW) {
      v = embw[(b * NW + r) * EMB + c];
    } else {
      int s = r - NW;
      v = (c < HID) ? cp[(b * NSENT + s) * HIDP + c]
                    : lf[(b * NSENT + s) * HIDP + (c - HID)];
    }
  }
  dst[gid] = (_Float16)v;
}

__global__ void k_deg_init(float* __restrict__ deg) {
  int gid = blockIdx.x * blockDim.x + threadIdx.x;
  if (gid < NNODE) deg[gid] = 1.f;                   // self loop
}

__global__ void k_deg_edges(const int* __restrict__ dst, float* __restrict__ deg) {
  int e = blockIdx.x * blockDim.x + threadIdx.x;
  if (e < EDGES) atomicAdd(&deg[dst[e]], 1.f);
}

__global__ void k_dis(const float* __restrict__ deg, float* __restrict__ dis) {
  int gid = blockIdx.x * blockDim.x + threadIdx.x;
  if (gid < NNODE) dis[gid] = rsqrtf(deg[gid]);      // deg >= 1 always
}

// agg[dst] += xw[src] * dis[src]*dis[dst] over all graph edges
__global__ void k_scatter(const int* __restrict__ src, const int* __restrict__ dst,
                          const float* __restrict__ dis, const float* __restrict__ xw,
                          float* __restrict__ agg) {
  int e = blockIdx.x * blockDim.x + threadIdx.x;
  if (e >= EDGES) return;
  int s = src[e], dd = dst[e];
  float nrm = dis[s] * dis[dd];
  const float* xr = xw + s * HIDP;
  float* ar = agg + dd * HIDP;
  __builtin_prefetch(xr, 0, 1);                      // -> global_prefetch_b8
  for (int c = 0; c < HID; ++c) atomicAdd(&ar[c], xr[c] * nrm);
}

// out = act(agg + xw*dis^2 (self loop) + bias)
__global__ void k_finalize(const float* __restrict__ agg, const float* __restrict__ xw,
                           const float* __restrict__ dis, const float* __restrict__ bias,
                           float* __restrict__ out, int outld, int do_elu) {
  int gid = blockIdx.x * blockDim.x + threadIdx.x;
  if (gid >= NNODE * HID) return;
  int i = gid / HID, c = gid - i * HID;
  float di = dis[i];
  float v = agg[i * HIDP + c] + xw[i * HIDP + c] * di * di + bias[c];
  if (do_elu) v = (v > 0.f) ? v : (__expf(v) - 1.f);
  out[i * outld + c] = v;
}

// ------------------------------- launcher -----------------------------------

#define LAUNCH(kern, n, ...) kern<<<dim3(CDIV((n), 256)), dim3(256), 0, stream>>>(__VA_ARGS__)
#define GEMM_GRID(M, Np) dim3(CDIV(((M) / 32) * ((Np) / 32), 8))

extern "C" void kernel_launch(void* const* d_in, const int* in_sizes, int n_in,
                              void* d_out, int out_size, void* d_ws, size_t ws_size,
                              hipStream_t stream) {
  (void)in_sizes; (void)n_in; (void)out_size; (void)ws_size;

  const int*   x        = (const int*)d_in[0];
  const int*   sent_x   = (const int*)d_in[1];
  const int*   ei       = (const int*)d_in[2];
  const float* etab     = (const float*)d_in[3];
  const float* conv_w[6], *conv_b[6];
  for (int h = 0; h < 6; ++h) { conv_w[h] = (const float*)d_in[4 + 2 * h];
                                conv_b[h] = (const float*)d_in[5 + 2 * h]; }
  const float* cnn_proj_w = (const float*)d_in[16];
  const float* cnn_proj_b = (const float*)d_in[17];
  const float* w_ih0 = (const float*)d_in[18];
  const float* w_hh0 = (const float*)d_in[19];
  const float* b_ih0 = (const float*)d_in[20];
  const float* b_hh0 = (const float*)d_in[21];
  const float* w_ih1 = (const float*)d_in[22];
  const float* w_hh1 = (const float*)d_in[23];
  const float* b_ih1 = (const float*)d_in[24];
  const float* b_hh1 = (const float*)d_in[25];
  const float* lproj_w = (const float*)d_in[26];
  const float* lproj_b = (const float*)d_in[27];
  const float* gc0_w = (const float*)d_in[28];
  const float* gc0_b = (const float*)d_in[29];
  const float* gc1_w = (const float*)d_in[30];
  const float* gc1_b = (const float*)d_in[31];
  const float* gc2_w = (const float*)d_in[32];
  const float* gc2_b = (const float*)d_in[33];

  float* out_gcn = (float*)d_out;                      // [NNODE, HID]
  float* out_emb = (float*)d_out + NNODE * HID;        // [BB, NW, EMB]

  // ---- bump allocator over d_ws ----
  char* base = (char*)d_ws;
  size_t off = 0;
  auto alloc = [&](size_t bytes) -> void* {
    void* p = base + off;
    off = (off + bytes + 255) & ~(size_t)255;
    return p;
  };

  _Float16* enc_h   = (_Float16*)alloc((size_t)(NN * SL * EMBP + 16384) * 2); // +slack for tile over-read
  _Float16* convW   = (_Float16*)alloc((size_t)27 * EMBP * 64 * 2);
  unsigned int* feats = (unsigned int*)alloc((size_t)NN * EMB * 4);
  _Float16* ngram_h = (_Float16*)alloc((size_t)NN * EMBP * 2);
  _Float16* wT_cnn  = (_Float16*)alloc((size_t)EMBP * HIDP * 2);
  float*    cp      = (float*)alloc((size_t)NN * HIDP * 4);         // cnn_feat
  _Float16* X0h     = (_Float16*)alloc((size_t)NN * HIDP * 2);
  _Float16* wT_ih0  = (_Float16*)alloc((size_t)2 * HIDP * G4P * 2);
  _Float16* wT_ih1  = (_Float16*)alloc((size_t)2 * EMBP * G4P * 2);
  float*    bsum0   = (float*)alloc((size_t)2 * G4 * 4);
  float*    bsum1   = (float*)alloc((size_t)2 * G4 * 4);
  float*    xpart   = (float*)alloc((size_t)2 * NN * G4P * 4);      // reused l0/l1
  float*    h1      = (float*)alloc((size_t)NN * EMBP * 4);
  _Float16* h1h     = (_Float16*)alloc((size_t)NN * EMBP * 2);
  float*    h2      = (float*)alloc((size_t)NN * EMBP * 4);
  _Float16* h2h     = (_Float16*)alloc((size_t)NN * EMBP * 2);
  _Float16* wT_lp   = (_Float16*)alloc((size_t)EMBP * HIDP * 2);
  float*    lf      = (float*)alloc((size_t)NN * HIDP * 4);         // lstm_feat
  _Float16* nodes_h = (_Float16*)alloc((size_t)NNODE * EMBP * 2);
  _Float16* wT_gc0  = (_Float16*)alloc((size_t)EMBP * HIDP * 2);
  float*    nodes0  = (float*)alloc((size_t)NNODE * HIDP * 4);
  float*    deg     = (float*)alloc((size_t)NNODE * 4);
  float*    dis     = (float*)alloc((size_t)NNODE * 4);
  _Float16* xin_h   = (_Float16*)alloc((size_t)NNODE * HIDP * 2);
  _Float16* wT_gc   = (_Float16*)alloc((size_t)HIDP * HIDP * 2);
  float*    xw      = (float*)alloc((size_t)NNODE * HIDP * 4);
  float*    agg     = (float*)alloc((size_t)NNODE * HIDP * 4);
  float*    hbuf    = (float*)alloc((size_t)NNODE * HIDP * 4);
  int*      sentlen = (int*)alloc((size_t)NN * 4);
  int*      glen    = (int*)alloc((size_t)BB * 4);

  const int* esrc = ei;
  const int* edst = ei + EDGES;

  // ---- 1. word-node embeddings (also part of the output tuple) ----
  LAUNCH(k_embed_words, BB * NW * EMB, x, etab, out_emb);

  // ---- 2. sentence token encodings + positional ----
  LAUNCH(k_lens, NN, sent_x, sentlen, glen);
  LAUNCH(k_enc, NN * SL * EMBP, sent_x, etab, sentlen, enc_h);

  // ---- 3. CNN n-gram features (6 window heights) ----
  hipMemsetAsync(feats, 0, (size_t)NN * EMB * 4, stream);
  size_t convOff[7];
  {
    size_t o = 0;
    for (int h = 2; h <= 7; ++h) { convOff[h - 2] = o; o += (size_t)h * EMBP * 64; }
  }
  for (int h = 2; h <= 7; ++h) {
    _Float16* wdst = convW + convOff[h - 2];
    LAUNCH(k_pack_convw, h * EMBP * 64, conv_w[h - 2], wdst, h);
    k_conv<<<dim3(CDIV(NN * 6, 8)), dim3(256), 0, stream>>>(enc_h, wdst, conv_b[h - 2], feats, h);
  }

  // ---- 4. cnn projection: cnn_feat = (ngram + pos) @ Wcnn^T + b ----
  LAUNCH(k_ngram, NN * EMBP, feats, ngram_h);
  LAUNCH(k_pack_wT, EMBP * HIDP, cnn_proj_w, wT_cnn, HID, EMB, EMBP, HIDP);
  k_wmma_gemm<<<GEMM_GRID(NN, HIDP), dim3(256), 0, stream>>>(
      ngram_h, wT_cnn, cp, cnn_proj_b, NN, EMBP, HIDP, HID);

  // ---- 5. LSTM layer 0 ----
  LAUNCH(k_pack_a, NN * HIDP, cp, X0h, NN, HID, HIDP, HIDP);
  LAUNCH(k_addbias, 2 * G4, b_ih0, b_hh0, bsum0, 2 * G4);
  for (int d = 0; d < 2; ++d) {
    LAUNCH(k_pack_wT, HIDP * G4P, w_ih0 + d * G4 * HID, wT_ih0 + d * HIDP * G4P,
           G4, HID, HIDP, G4P);
    k_wmma_gemm<<<GEMM_GRID(NN, G4P), dim3(256), 0, stream>>>(
        X0h, wT_ih0 + d * HIDP * G4P, xpart + d * NN * G4P, bsum0 + d * G4, NN, HIDP, G4P, G4);
  }
  k_lstm<<<dim3(BB, 2), dim3(640), 0, stream>>>(xpart, w_hh0, glen, h1);

  // ---- 6. LSTM layer 1 ----
  LAUNCH(k_pack_a, NN * EMBP, h1, h1h, NN, 2 * HID, EMBP, EMBP);
  LAUNCH(k_addbias, 2 * G4, b_ih1, b_hh1, bsum1, 2 * G4);
  for (int d = 0; d < 2; ++d) {
    LAUNCH(k_pack_wT, EMBP * G4P, w_ih1 + d * G4 * 2 * HID, wT_ih1 + d * EMBP * G4P,
           G4, 2 * HID, EMBP, G4P);
    k_wmma_gemm<<<GEMM_GRID(NN, G4P), dim3(256), 0, stream>>>(
        h1h, wT_ih1 + d * EMBP * G4P, xpart + d * NN * G4P, bsum1 + d * G4, NN, EMBP, G4P, G4);
  }
  k_lstm<<<dim3(BB, 2), dim3(640), 0, stream>>>(xpart, w_hh1, glen, h2);

  // ---- 7. lstm projection (masked) ----
  LAUNCH(k_pack_mask, NN * EMBP, h2, glen, h2h);
  LAUNCH(k_pack_wT, EMBP * HIDP, lproj_w, wT_lp, HID, 2 * HID, EMBP, HIDP);
  k_wmma_gemm<<<GEMM_GRID(NN, HIDP), dim3(256), 0, stream>>>(
      h2h, wT_lp, lf, lproj_b, NN, EMBP, HIDP, HID);

  // ---- 8. node features + gc0 ----
  LAUNCH(k_nodes, NNODE * EMBP, out_emb, cp, lf, nodes_h);
  LAUNCH(k_pack_wT, EMBP * HIDP, gc0_w, wT_gc0, HID, EMB, EMBP, HIDP);
  k_wmma_gemm<<<GEMM_GRID(NNODE, HIDP), dim3(256), 0, stream>>>(
      nodes_h, wT_gc0, nodes0, gc0_b, NNODE, EMBP, HIDP, HID);

  // ---- 9. GCN normalization ----
  LAUNCH(k_deg_init, NNODE, deg);
  LAUNCH(k_deg_edges, EDGES, edst, deg);
  LAUNCH(k_dis, NNODE, deg, dis);

  // ---- 10. GCN layer 1 (elu) ----
  LAUNCH(k_pack_a, NNODE * HIDP, nodes0, xin_h, NNODE, HID, HIDP, HIDP);
  LAUNCH(k_pack_wT, HIDP * HIDP, gc1_w, wT_gc, HID, HID, HIDP, HIDP);
  k_wmma_gemm<<<GEMM_GRID(NNODE, HIDP), dim3(256), 0, stream>>>(
      xin_h, wT_gc, xw, nullptr, NNODE, HIDP, HIDP, HID);
  hipMemsetAsync(agg, 0, (size_t)NNODE * HIDP * 4, stream);
  LAUNCH(k_scatter, EDGES, esrc, edst, dis, xw, agg);
  LAUNCH(k_finalize, NNODE * HID, agg, xw, dis, gc1_b, hbuf, HIDP, 1);

  // ---- 11. GCN layer 2 (output) ----
  LAUNCH(k_pack_a, NNODE * HIDP, hbuf, xin_h, NNODE, HID, HIDP, HIDP);
  LAUNCH(k_pack_wT, HIDP * HIDP, gc2_w, wT_gc, HID, HID, HIDP, HIDP);
  k_wmma_gemm<<<GEMM_GRID(NNODE, HIDP), dim3(256), 0, stream>>>(
      xin_h, wT_gc, xw, nullptr, NNODE, HIDP, HIDP, HID);
  hipMemsetAsync(agg, 0, (size_t)NNODE * HIDP * 4, stream);
  LAUNCH(k_scatter, EDGES, esrc, edst, dis, xw, agg);
  LAUNCH(k_finalize, NNODE * HID, agg, xw, dis, gc2_b, out_gcn, HID, 0);
}
